// MessgePsssing_4861902979531
// MI455X (gfx1250) — matrix operand order repaired and checked by reading, not compiled, not run on hardware
//
#include <hip/hip_runtime.h>

// ---------------------------------------------------------------------------
// GCN layer: out = (A / rowsum|A|) @ X @ W + bias
// Restructured as: Y = X@W (split to bf16 hi/lo, transposed);
//                  P_z = A[:, Kz] @ Y (3-term bf16 WMMA, fused |A| rowsums);
//                  out = (sum_z P_z) / max(rowsum, eps) + bias
// All B-tile / lo-plane addressing uses compile-time offsets from a single
// moving base pointer so they fold into the load IOFFSET immediate: keeps the
// loads on the GLOBAL path (no flat/DScnt), eliminates per-tile address VALU,
// and avoids WMMA source-register WAR hazards (ISA 7.12.1).
// ---------------------------------------------------------------------------

typedef __attribute__((ext_vector_type(16))) __bf16 v16bf;
typedef __attribute__((ext_vector_type(8)))  float  v8f;
typedef __attribute__((ext_vector_type(4)))  float  v4f;

#define N_NODES 8192
#define F_IN    512
#define F_OUT   256
#define KSPLIT  4
#define KCHUNK  (N_NODES / KSPLIT)
#define GCN_EPS 1e-12f

// Element offsets of the "lo" planes inside the packed hi|lo buffers.
#define YT_LO  ((size_t)F_OUT * N_NODES)   // 2 MB elems -> +4 MB bytes
#define WT_LO  ((size_t)F_IN * F_OUT)      // 128 K elems -> +256 KB bytes

#define WMMA_BF16(A, B, C) \
    __builtin_amdgcn_wmma_f32_16x16x32_bf16(false, (A), false, (B), (short)0, (C), false, false)

__device__ __forceinline__ void splitbf(float x, __bf16& h, __bf16& l) {
    h = (__bf16)x;                 // hi part (RNE)
    l = (__bf16)(x - (float)h);    // residual
}

// ---------------------------------------------------------------------------
// Kernel 0: transpose + hi/lo split of W into packed Wt: hi[n][k] | lo[n][k]
// ---------------------------------------------------------------------------
__global__ void k_prep_w(const float* __restrict__ W, __bf16* __restrict__ Wt) {
    int t = blockIdx.x * blockDim.x + threadIdx.x;
    if (t < F_IN * F_OUT) {
        int k = t / F_OUT, n = t % F_OUT;
        __bf16 h, l;
        splitbf(W[t], h, l);
        Wt[(size_t)n * F_IN + k]         = h;
        Wt[WT_LO + (size_t)n * F_IN + k] = l;
    }
}

// Build A-fragment (16x32 bf16, contiguous k = half*16 + s, same as B) + hi/lo split from a lane
// pointer already offset by half*16; `o` is a compile-time element offset so
// the loads fold into the instruction IOFFSET. Returns sum|a| of the 16 vals.
__device__ __forceinline__ float load_a_frag(const float* __restrict__ aptr,
                                             int o, v16bf& ah, v16bf& al) {
    v4f p0 = *(const v4f*)(aptr + o);
    v4f p1 = *(const v4f*)(aptr + o + 4);
    v4f p2 = *(const v4f*)(aptr + o + 8);
    v4f p3 = *(const v4f*)(aptr + o + 12);
    float sabs = 0.0f;
#pragma unroll
    for (int i = 0; i < 4; ++i) {
        __bf16 h, l;
        splitbf(p0[i], h, l); ah[i]      = h; al[i]      = l;
        splitbf(p1[i], h, l); ah[4 + i]  = h; al[4 + i]  = l;
        splitbf(p2[i], h, l); ah[8 + i]  = h; al[8 + i]  = l;
        splitbf(p3[i], h, l); ah[12 + i] = h; al[12 + i] = l;
        sabs += __builtin_fabsf(p0[i]) + __builtin_fabsf(p1[i])
              + __builtin_fabsf(p2[i]) + __builtin_fabsf(p3[i]);
    }
    return sabs;
}

// ---------------------------------------------------------------------------
// Kernel 1: Y = X @ W via 3-term bf16-split WMMA.
// One wave computes a 16(M) x 64(N) tile. Output written transposed,
// pre-split to bf16 hi/lo planes of Yt: Yt[n][row] | Yt[YT_LO + n][row]
// (row-contiguous == K-contiguous for the next GEMM's B operand).
// ---------------------------------------------------------------------------
__global__ void __launch_bounds__(32)
k_xw(const float* __restrict__ X, const __bf16* __restrict__ Wt,
     __bf16* __restrict__ Yt) {
    const int lane = threadIdx.x;
    const int half = lane >> 4;      // 0: lanes 0-15, 1: lanes 16-31
    const int mrow = lane & 15;
    const int rowBase = blockIdx.x * 16;
    const int nBase   = blockIdx.y * 64;

    v8f zero = {};
    v8f acc[4];
#pragma unroll
    for (int j = 0; j < 4; ++j) acc[j] = zero;

    const float*  aptr = X + (size_t)(rowBase + mrow) * F_IN + half * 16;
    const __bf16* bptr = Wt + (size_t)(nBase + mrow) * F_IN + half * 16;
    const size_t  JS   = (size_t)16 * F_IN;   // N-tile stride (elements)

    for (int k0 = 0; k0 < F_IN; k0 += 32) {
        v16bf ah, al;
        (void)load_a_frag(aptr, 0, ah, al);

#pragma unroll
        for (int j = 0; j < 4; j += 2) {
            v16bf b0h = *(const v16bf*)(bptr + (size_t)j * JS);
            v16bf b0l = *(const v16bf*)(bptr + (size_t)j * JS + WT_LO);
            v16bf b1h = *(const v16bf*)(bptr + (size_t)(j + 1) * JS);
            v16bf b1l = *(const v16bf*)(bptr + (size_t)(j + 1) * JS + WT_LO);
            acc[j]     = WMMA_BF16(ah, b0h, acc[j]);
            acc[j + 1] = WMMA_BF16(ah, b1h, acc[j + 1]);
            acc[j]     = WMMA_BF16(ah, b0l, acc[j]);
            acc[j + 1] = WMMA_BF16(ah, b1l, acc[j + 1]);
            acc[j]     = WMMA_BF16(al, b0h, acc[j]);
            acc[j + 1] = WMMA_BF16(al, b1h, acc[j + 1]);
        }
        aptr += 32;
        bptr += 32;
    }

    // C/D layout: VGPR e -> row rowBase + e + half*8, col = nTile + mrow.
#pragma unroll
    for (int j = 0; j < 4; ++j) {
        int n = nBase + j * 16 + mrow;
#pragma unroll
        for (int e = 0; e < 8; ++e) {
            int row = rowBase + e + half * 8;
            __bf16 h, l;
            splitbf(acc[j][e], h, l);
            Yt[(size_t)n * N_NODES + row]         = h;
            Yt[YT_LO + (size_t)n * N_NODES + row] = l;
        }
    }
}

// ---------------------------------------------------------------------------
// Kernel 2: P_z = A[:, z-chunk] @ Y  (the 268 MB pass; adjacency read twice
// total across the 2 N-chunks). Fuses |A| row-sum partials (deterministic,
// no atomics: only blockIdx.y==0 waves store them).
// One wave: 16(M) x 128(N), K chunk of 2048, unrolled 2x32 elements.
// All 32 B-tile loads per step use immediate offsets from one base pointer
// (tile stride 256 KB, lo plane +4 MB; both < 8.38 MB signed IOFFSET).
// ---------------------------------------------------------------------------
__global__ void __launch_bounds__(32)
k_spmm(const float* __restrict__ adj, const __bf16* __restrict__ Yt,
       float* __restrict__ partial, float* __restrict__ rspart) {
    const int lane = threadIdx.x;
    const int half = lane >> 4;
    const int mrow = lane & 15;
    const int rowBase = blockIdx.x * 16;
    const int nBase   = blockIdx.y * 128;
    const int z       = blockIdx.z;
    const int kStart  = z * KCHUNK;

    v8f zero = {};
    v8f acc[8];
#pragma unroll
    for (int j = 0; j < 8; ++j) acc[j] = zero;

    float sabs = 0.0f;
    const float*  arow0 = adj + (size_t)(rowBase + mrow) * N_NODES + half * 16;
    const float*  aptr  = arow0 + kStart;
    const __bf16* bptr  = Yt + (size_t)(nBase + mrow) * N_NODES + kStart + half * 16;
    const size_t  JS    = (size_t)16 * N_NODES;   // N-tile stride (elements)

    for (int kk = 0; kk < KCHUNK; kk += 64) {
        // Stream-prefetch adjacency ~1KB ahead, wrapped within the row so the
        // address never leaves the allocation. Lowers to global_prefetch_b8.
        __builtin_prefetch(arow0 + ((kStart + kk + 256) & (N_NODES - 1)), 0, 0);

#pragma unroll
        for (int u = 0; u < 2; ++u) {
            const int o = u * 32;   // compile-time: folds into load IOFFSET
            v16bf ah, al;
            sabs += load_a_frag(aptr, o, ah, al);

#pragma unroll
            for (int j = 0; j < 8; j += 2) {
                v16bf b0h = *(const v16bf*)(bptr + (size_t)j * JS + o);
                v16bf b0l = *(const v16bf*)(bptr + (size_t)j * JS + YT_LO + o);
                v16bf b1h = *(const v16bf*)(bptr + (size_t)(j + 1) * JS + o);
                v16bf b1l = *(const v16bf*)(bptr + (size_t)(j + 1) * JS + YT_LO + o);
                // Alternate accumulators: no consecutive WMMAs share D.
                acc[j]     = WMMA_BF16(ah, b0h, acc[j]);
                acc[j + 1] = WMMA_BF16(ah, b1h, acc[j + 1]);
                acc[j]     = WMMA_BF16(ah, b0l, acc[j]);
                acc[j + 1] = WMMA_BF16(ah, b1l, acc[j + 1]);
                acc[j]     = WMMA_BF16(al, b0h, acc[j]);
                acc[j + 1] = WMMA_BF16(al, b1h, acc[j + 1]);
            }
        }

        aptr += 64;
        bptr += 64;
    }

    // Store f32 partial tile: partial[z][row][col]
#pragma unroll
    for (int j = 0; j < 8; ++j) {
        int col = nBase + j * 16 + mrow;
#pragma unroll
        for (int e = 0; e < 8; ++e) {
            int row = rowBase + e + half * 8;
            partial[((size_t)z * N_NODES + row) * F_OUT + col] = acc[j][e];
        }
    }

    // Fused L1 row-sum partial for this K chunk (each k read exactly once per
    // row across the two lane halves). Only one N-chunk contributes.
    if (blockIdx.y == 0) {
        float other = __shfl_down(sabs, 16, 32);   // lane m += lane m+16
        if (lane < 16)
            rspart[(size_t)z * N_NODES + rowBase + mrow] = sabs + other;
    }
}

// ---------------------------------------------------------------------------
// Kernel 3: out[row][n] = (sum_z P_z[row][n]) / max(rowsum[row], eps) + bias[n]
// ---------------------------------------------------------------------------
__global__ void __launch_bounds__(F_OUT)
k_finish(const float* __restrict__ partial, const float* __restrict__ rspart,
         const float* __restrict__ bias, float* __restrict__ out) {
    const int row = blockIdx.x;
    const int n   = threadIdx.x;

    float rs = 0.0f;
#pragma unroll
    for (int zi = 0; zi < KSPLIT; ++zi) rs += rspart[(size_t)zi * N_NODES + row];
    const float inv = 1.0f / fmaxf(rs, GCN_EPS);

    float s = 0.0f;
#pragma unroll
    for (int zi = 0; zi < KSPLIT; ++zi)
        s += partial[((size_t)zi * N_NODES + row) * F_OUT + n];

    out[(size_t)row * F_OUT + n] = s * inv + bias[n];
}

// ---------------------------------------------------------------------------
extern "C" void kernel_launch(void* const* d_in, const int* in_sizes, int n_in,
                              void* d_out, int out_size, void* d_ws, size_t ws_size,
                              hipStream_t stream) {
    const float* adj  = (const float*)d_in[0];   // [8192, 8192]
    const float* X    = (const float*)d_in[1];   // [8192, 512]
    const float* W    = (const float*)d_in[2];   // [512, 256]
    const float* bias = (const float*)d_in[3];   // [256]
    float* out = (float*)d_out;                  // [8192, 256]

    // Workspace carve-out (256B aligned slices)
    char* p = (char*)d_ws;
    auto carve = [&p](size_t bytes) -> char* {
        char* r = p;
        p += (bytes + 255) & ~(size_t)255;
        return r;
    };
    // Packed hi|lo buffers (lo plane at a compile-time element offset).
    __bf16* Wt = (__bf16*)carve(2 * (size_t)F_IN * F_OUT * sizeof(__bf16));
    __bf16* Yt = (__bf16*)carve(2 * (size_t)F_OUT * N_NODES * sizeof(__bf16));
    float* partial = (float*)carve((size_t)KSPLIT * N_NODES * F_OUT * sizeof(float));
    float* rspart  = (float*)carve((size_t)KSPLIT * N_NODES * sizeof(float));

    k_prep_w<<<(F_IN * F_OUT + 255) / 256, 256, 0, stream>>>(W, Wt);

    k_xw<<<dim3(N_NODES / 16, F_OUT / 64), 32, 0, stream>>>(X, Wt, Yt);

    k_spmm<<<dim3(N_NODES / 16, 2, KSPLIT), 32, 0, stream>>>(adj, Yt, partial, rspart);

    k_finish<<<N_NODES, F_OUT, 0, stream>>>(partial, rspart, bias, out);
}
